// MultiHeadAttentionLayer_21371757265248
// MI455X (gfx1250) — matrix operand order repaired and checked by reading, not compiled, not run on hardware
//
#include <hip/hip_runtime.h>
#include <hip/hip_bf16.h>

typedef __bf16 bf16x16 __attribute__((ext_vector_type(16)));
typedef float f32x8  __attribute__((ext_vector_type(8)));

union BF16Frag { bf16x16 v; uint4 q[2]; };
union Pack4    { __bf16 h[4]; uint2 u; };

#define IN_DIM 512
#define NODES  128
#define HEADS  8
#define HDIM   64

// ---------------------------------------------------------------------------
// bf16 WMMA micro-kernel: one wave computes a 16(M) x 64(N) tile,
// K = 512 in steps of 32 via v_wmma_f32_16x16x32_bf16.
// ---------------------------------------------------------------------------
__device__ __forceinline__ void wmma_gemm_16x64_k512(
    const __bf16* __restrict__ aBase,
    const __bf16* __restrict__ bBase,
    int ln, int half, f32x8 acc[4])
{
#pragma unroll 4
    for (int k0 = 0; k0 < IN_DIM; k0 += 32) {
        BF16Frag a;
        const __bf16* ap = aBase + ln * IN_DIM + k0 + half * 8;
        a.q[0] = *(const uint4*)(ap);        // K = k0+off   .. +8
        a.q[1] = *(const uint4*)(ap + 16);   // K = k0+16+off.. +8
#pragma unroll
        for (int t = 0; t < 4; ++t) {
            BF16Frag b;
            const __bf16* bp = bBase + (t * 16 + ln) * IN_DIM + k0 + half * 8;
            b.q[0] = *(const uint4*)(bp);
            b.q[1] = *(const uint4*)(bp + 16);
            acc[t] = __builtin_amdgcn_wmma_f32_16x16x32_bf16(
                false, a.v, false, b.v, (short)0, acc[t], false, false);
        }
    }
}

// ---------------------------------------------------------------------------
// Async global->LDS copy helpers (ASYNCcnt-tracked, GVS addressing)
// ---------------------------------------------------------------------------
__device__ __forceinline__ void async_b128(void* ldsPtr, const void* sBase, int byteOff)
{
    unsigned ldsAddr = (unsigned)(uintptr_t)ldsPtr;   // LDS offset in low 32 bits
    asm volatile("global_load_async_to_lds_b128 %0, %1, %2"
                 :: "v"(ldsAddr), "v"(byteOff), "s"(sBase) : "memory");
}
__device__ __forceinline__ void wait_async0()
{
    asm volatile("s_wait_asynccnt 0x0" ::: "memory");
}

// Issue one head's We slice (64KB contiguous) + K slice (128 x 64 bf16,
// row stride 512) into the given double-buffer halves.
__device__ __forceinline__ void issue_head_loads(
    int tid, const __bf16* __restrict__ WeT, const __bf16* __restrict__ KwsB,
    int h, __bf16* wbuf, __bf16* kbuf)
{
    const char* wbase = (const char*)(WeT + (size_t)h * HDIM * IN_DIM);
    char* wd = (char*)wbuf;
#pragma unroll
    for (int it = 0; it < 16; ++it) {
        int id = tid + 256 * it;                 // 4096 x 16B = 64 KB
        async_b128(wd + id * 16, wbase, id * 16);
    }
    const char* kbase = (const char*)(KwsB + h * HDIM);
    char* kd = (char*)kbuf;
#pragma unroll
    for (int it = 0; it < 4; ++it) {
        int id = tid + 256 * it;                 // 1024 x 16B = 16 KB
        int j = id >> 3, cc = id & 7;
        async_b128(kd + id * 16, kbase, j * 1024 + cc * 16);
    }
}

// ---------------------------------------------------------------------------
// Transpose + convert weights to bf16: WT[o][c] = (bf16)W[c][o]
// ---------------------------------------------------------------------------
__global__ __launch_bounds__(256) void prep_weights(
    const float* __restrict__ Wq, const float* __restrict__ Wk,
    const float* __restrict__ We, const float* __restrict__ Wv,
    __bf16* __restrict__ wt)
{
    const float* src = (blockIdx.y == 0) ? Wq :
                       (blockIdx.y == 1) ? Wk :
                       (blockIdx.y == 2) ? We : Wv;
    __bf16* dst = wt + (size_t)blockIdx.y * (IN_DIM * IN_DIM);
    int idx = blockIdx.x * 256 + threadIdx.x;
    int o = idx >> 9, c = idx & 511;
    dst[idx] = (__bf16)src[c * IN_DIM + o];
}

// ---------------------------------------------------------------------------
// Q/K/V projections. grid = 8 blocks, each handles 128 rows of h (1024x512).
// ---------------------------------------------------------------------------
__global__ __launch_bounds__(256) void qkv_proj(
    const float* __restrict__ hin, const __bf16* __restrict__ wt,
    const float* __restrict__ bq, const float* __restrict__ bk,
    const float* __restrict__ bv,
    float* __restrict__ Qws, __bf16* __restrict__ Kws, float* __restrict__ Vws)
{
    extern __shared__ char smem[];
    __bf16* xLds = (__bf16*)smem;                 // 131072 B
    __bf16* wLds = (__bf16*)(smem + 131072);      //  65536 B

    const int tid = threadIdx.x;
    const int rowblk = blockIdx.x * 128;

    {
        const float4* src4 = (const float4*)(hin + (size_t)rowblk * IN_DIM);
        uint2* dst2 = (uint2*)xLds;
#pragma unroll 4
        for (int it = 0; it < 64; ++it) {
            int c = tid + 256 * it;
            float4 f = src4[c];
            Pack4 p;
            p.h[0] = (__bf16)f.x; p.h[1] = (__bf16)f.y;
            p.h[2] = (__bf16)f.z; p.h[3] = (__bf16)f.w;
            dst2[c] = p.u;
        }
    }

    const int wave = tid >> 5, lane = tid & 31, ln = lane & 15, half = lane >> 4;

    for (int mat = 0; mat < 3; ++mat) {
        const __bf16* WT = wt + (size_t)((mat == 0) ? 0 : (mat == 1) ? 1 : 3) *
                                (IN_DIM * IN_DIM);          // q, k, v
        const float* bias = (mat == 0) ? bq : (mat == 1) ? bk : bv;

        for (int o0 = 0; o0 < IN_DIM; o0 += 64) {
            __syncthreads();
            {
                const uint4* ws4 = (const uint4*)(WT + (size_t)o0 * IN_DIM);
                uint4* wd4 = (uint4*)wLds;
#pragma unroll 4
                for (int it = 0; it < 16; ++it) wd4[tid + 256 * it] = ws4[tid + 256 * it];
            }
            __syncthreads();

            f32x8 acc[4];
#pragma unroll
            for (int t = 0; t < 4; ++t) {
                float bb = bias[o0 + t * 16 + ln];
#pragma unroll
                for (int v = 0; v < 8; ++v) acc[t][v] = bb;
            }
            wmma_gemm_16x64_k512(xLds + (wave * 16) * IN_DIM, wLds, ln, half, acc);

#pragma unroll
            for (int t = 0; t < 4; ++t) {
                int col = o0 + t * 16 + ln;
#pragma unroll
                for (int v = 0; v < 8; ++v) {
                    int grow = rowblk + wave * 16 + v + half * 8;
                    float val = acc[t][v];
                    if (mat == 0)      Qws[(size_t)grow * IN_DIM + col] = val;
                    else if (mat == 1) Kws[(size_t)grow * IN_DIM + col] = (__bf16)(val * 0.125f);
                    else               Vws[(size_t)grow * IN_DIM + col] = val;
                }
            }
        }
    }
}

// ---------------------------------------------------------------------------
// Fused edge-GEMM + triple-product scores + exp/kRW + normalize + S@V.
// One workgroup per (b,i); async double-buffered W/K head slices.
// ---------------------------------------------------------------------------
__global__ __launch_bounds__(256) void attn_main(
    const float* __restrict__ ein, const float* __restrict__ kRW,
    const __bf16* __restrict__ WeT, const float* __restrict__ be,
    const float* __restrict__ Qws, const __bf16* __restrict__ Kws,
    const float* __restrict__ Vws, float* __restrict__ out)
{
    extern __shared__ char smem[];
    __bf16* eLds  = (__bf16*)smem;               // 131072 B
    __bf16* wLds0 = (__bf16*)(smem + 131072);    //  65536 B
    __bf16* wLds1 = (__bf16*)(smem + 196608);    //  65536 B
    __bf16* kLds0 = (__bf16*)(smem + 262144);    //  16384 B
    __bf16* kLds1 = (__bf16*)(smem + 278528);    //  16384 B
    float*  qLds  = (float*)(smem + 294912);     //   2048 B
    float*  sLds  = (float*)(smem + 296960);     //   4096 B
    float*  dLds  = (float*)(smem + 301056);     //     32 B

    const int tid = threadIdx.x;
    const int bi = blockIdx.x;                  // b*128 + i
    const int b  = bi >> 7;
    const __bf16* KwsB = Kws + (size_t)b * NODES * IN_DIM;

    // kick off head-0 W/K slices asynchronously (overlaps the e-tile convert)
    issue_head_loads(tid, WeT, KwsB, 0, wLds0, kLds0);

    // load + convert e tile (this (b,i): all j rows)
    {
        const float4* src4 = (const float4*)(ein + (size_t)bi * NODES * IN_DIM);
        uint2* dst2 = (uint2*)eLds;
#pragma unroll 4
        for (int it = 0; it < 64; ++it) {
            int c = tid + 256 * it;
            float4 f = src4[c];
            Pack4 p;
            p.h[0] = (__bf16)f.x; p.h[1] = (__bf16)f.y;
            p.h[2] = (__bf16)f.z; p.h[3] = (__bf16)f.w;
            dst2[c] = p.u;
        }
    }
    // load Q row (fp32)
    qLds[tid]       = Qws[(size_t)bi * IN_DIM + tid];
    qLds[tid + 256] = Qws[(size_t)bi * IN_DIM + tid + 256];

    const int wave = tid >> 5, lane = tid & 31, ln = lane & 15, half = lane >> 4;

    for (int h = 0; h < HEADS; ++h) {
        __bf16* wcur = (h & 1) ? wLds1 : wLds0;
        __bf16* kcur = (h & 1) ? kLds1 : kLds0;

        wait_async0();       // my async loads for this head's buffers are in LDS
        __syncthreads();     // everyone's loads landed; prior reads of next buf done

        if (h + 1 < HEADS)   // prefetch next head's slices into the other buffer
            issue_head_loads(tid, WeT, KwsB, h + 1,
                             (h & 1) ? wLds0 : wLds1, (h & 1) ? kLds0 : kLds1);

        // E tile for this head: rows j in [16w,16w+16), cols d in [0,64)
        f32x8 acc[4];
#pragma unroll
        for (int t = 0; t < 4; ++t) {
            float bev = be[h * HDIM + t * 16 + ln];
#pragma unroll
            for (int v = 0; v < 8; ++v) acc[t][v] = bev;     // + be fused in C
        }
        wmma_gemm_16x64_k512(eLds + (wave * 16) * IN_DIM, wcur, ln, half, acc);

        // scores[h][j] = sum_d Q[h,d] * K[j,h,d] * E[j,h,d]
        float rsum[8];
#pragma unroll
        for (int v = 0; v < 8; ++v) rsum[v] = 0.f;
#pragma unroll
        for (int t = 0; t < 4; ++t) {
            float qv = qLds[h * HDIM + t * 16 + ln];
#pragma unroll
            for (int v = 0; v < 8; ++v) {
                int j = wave * 16 + v + half * 8;
                float kv = (float)kcur[j * HDIM + t * 16 + ln];
                rsum[v] += acc[t][v] * qv * kv;
            }
        }
#pragma unroll
        for (int v = 0; v < 8; ++v) {       // reduce over the 16 d-lanes
            rsum[v] += __shfl_xor(rsum[v], 1, 32);
            rsum[v] += __shfl_xor(rsum[v], 2, 32);
            rsum[v] += __shfl_xor(rsum[v], 4, 32);
            rsum[v] += __shfl_xor(rsum[v], 8, 32);
        }
        if (ln == 0) {
#pragma unroll
            for (int v = 0; v < 8; ++v)
                sLds[h * NODES + wave * 16 + v + half * 8] = rsum[v];
        }
    }

    __syncthreads();
    if (tid < 8) dLds[tid] = 0.f;
    __syncthreads();
    {   // exp(clip) * k_RW, accumulate per-head denominator
        int hh = tid >> 5;
        const float* krw = kRW + (size_t)bi * NODES;
        float lsum = 0.f;
#pragma unroll
        for (int q = 0; q < 4; ++q) {
            int idx = tid * 4 + q;
            int j = idx & 127;
            float s = sLds[idx];
            s = fminf(fmaxf(s, -5.f), 5.f);
            s = __expf(s) * krw[j];
            sLds[idx] = s;
            lsum += s;
        }
        atomicAdd(&dLds[hh], lsum);
    }
    __syncthreads();

    // out[b,i,h,d] = (sum_j S[h,j] * V[b,j,h,d]) / max(denom[h], 1e-6)
    const float* Vb = Vws + (size_t)b * NODES * IN_DIM;
#pragma unroll
    for (int rep = 0; rep < 2; ++rep) {
        int hd = tid + rep * 256;
        int hh = hd >> 6;
        const float* srow = sLds + hh * NODES;
        float accv = 0.f;
#pragma unroll 4
        for (int j = 0; j < NODES; ++j)
            accv += srow[j] * Vb[(size_t)j * IN_DIM + hd];
        out[(size_t)bi * IN_DIM + hd] = accv / fmaxf(dLds[hh], 1e-6f);
    }
}

// ---------------------------------------------------------------------------
extern "C" void kernel_launch(void* const* d_in, const int* in_sizes, int n_in,
                              void* d_out, int out_size, void* d_ws, size_t ws_size,
                              hipStream_t stream) {
    (void)in_sizes; (void)n_in; (void)out_size; (void)ws_size;
    const float* h_   = (const float*)d_in[0];
    const float* e_   = (const float*)d_in[1];
    const float* kRW  = (const float*)d_in[2];
    const float* Wq   = (const float*)d_in[3];
    const float* bq   = (const float*)d_in[4];
    const float* Wk   = (const float*)d_in[5];
    const float* bk   = (const float*)d_in[6];
    const float* We   = (const float*)d_in[7];
    const float* be   = (const float*)d_in[8];
    const float* Wv   = (const float*)d_in[9];
    const float* bv   = (const float*)d_in[10];
    float* out = (float*)d_out;

    char* ws = (char*)d_ws;
    __bf16* wt  = (__bf16*)ws;                       // 4 x 512x512 bf16 = 2 MB
    float*  Qws = (float*)(ws + 2097152);            // 2 MB
    __bf16* Kws = (__bf16*)(ws + 4194304);           // 1 MB (pre-scaled bf16)
    float*  Vws = (float*)(ws + 5242880);            // 2 MB

    (void)hipFuncSetAttribute((const void*)qkv_proj,
        hipFuncAttributeMaxDynamicSharedMemorySize, 196608);
    (void)hipFuncSetAttribute((const void*)attn_main,
        hipFuncAttributeMaxDynamicSharedMemorySize, 301088);

    prep_weights<<<dim3(1024, 4), 256, 0, stream>>>(Wq, Wk, We, Wv, wt);
    qkv_proj<<<8, 256, 196608, stream>>>(h_, wt, bq, bk, bv, Qws, Kws, Vws);
    attn_main<<<1024, 256, 301088, stream>>>(
        e_, kRW, wt + (size_t)2 * IN_DIM * IN_DIM, be, Qws, Kws, Vws, out);
}
